// AttentionBlock_86517821215540
// MI455X (gfx1250) — compile-verified, matrix operand order checked by reading, not run on hardware
//
#include <hip/hip_runtime.h>
#include <hip/hip_bf16.h>

// ---------------------------------------------------------------------------
// Problem constants (from the reference): B=8, T=2048, D=1024, K=1024.
// ---------------------------------------------------------------------------
constexpr int Bc = 8;
constexpr int Tc = 2048;
constexpr int Dc = 1024;   // model dim
constexpr int Kc = 1024;   // projection dim (== Dc here)

typedef __attribute__((ext_vector_type(16))) __bf16 v16bf;
typedef __attribute__((ext_vector_type(8)))  float  v8f;

// float -> bf16 bits, round-to-nearest-even
__device__ __forceinline__ unsigned short f2bf(float x) {
  union { float f; unsigned u; } c; c.f = x;
  unsigned r = c.u + 0x7FFFu + ((c.u >> 16) & 1u);
  return (unsigned short)(r >> 16);
}

// 16-bit A/B fragment load for v_wmma_*_16x16x32_bf16.
// Caller passes rowBase = base + (lane&15)*ld + k0.  Per the CDNA5 ISA layout,
// lanes 0-15 hold K = {0..7, 16..23}, lanes 16-31 hold K = {8..15, 24..31}:
// two contiguous 16-byte runs per lane.
__device__ __forceinline__ v16bf load_frag16(const unsigned short* rowBase, int lane) {
  const unsigned short* p = rowBase + ((lane >> 4) & 1) * 8;
  union { v16bf v; uint4 u[2]; } f;
  f.u[0] = *reinterpret_cast<const uint4*>(p);
  f.u[1] = *reinterpret_cast<const uint4*>(p + 16);
  return f.v;
}

__device__ __forceinline__ v8f wmma_bf16(v16bf a, v16bf b, v8f c) {
  // (neg_a, A, neg_b, B, c_mod, C, reuse_a, reuse_b)
  return __builtin_amdgcn_wmma_f32_16x16x32_bf16(false, a, false, b, (short)0, c,
                                                 false, false);
}

// Async global->LDS copy (CDNA5 ASYNCcnt path).  Generic shared pointers on
// AMDGPU carry the LDS byte offset in their low 32 bits
// so truncation gives the VDST LDS address operand.
__device__ __forceinline__ void async_copy_b128(void* lds_ptr, const void* gptr) {
#if defined(__gfx1250__)
  unsigned lds_off = (unsigned)(unsigned long long)lds_ptr;
  asm volatile("global_load_async_to_lds_b128 %0, %1, off"
               :: "v"(lds_off), "v"(gptr) : "memory");
#else
  (void)lds_ptr; (void)gptr;
#endif
}

__device__ __forceinline__ void wait_asynccnt0() {
#if defined(__gfx1250__)
#if __has_builtin(__builtin_amdgcn_s_wait_asynccnt)
  __builtin_amdgcn_s_wait_asynccnt(0);
#else
  asm volatile("s_wait_asynccnt 0x0" ::: "memory");
#endif
#endif
}

// ---------------------------------------------------------------------------
// K0: elementwise fp32 -> bf16 cast
// ---------------------------------------------------------------------------
__global__ void cast_bf16_kernel(const float* __restrict__ src,
                                 unsigned short* __restrict__ dst, int n) {
  int i = blockIdx.x * blockDim.x + threadIdx.x;
  if (i < n) dst[i] = f2bf(src[i]);
}

// K0b: transpose-cast a 1024x1024 weight: Wt[n][d] = bf16(W[d][n])
__global__ void transpose_cast_kernel(const float* __restrict__ src,
                                      unsigned short* __restrict__ dst) {
  int i = blockIdx.x * blockDim.x + threadIdx.x;   // over Dc*Kc
  int d = i / Kc, n = i % Kc;
  dst[(size_t)n * Dc + d] = f2bf(src[(size_t)d * Kc + n]);
}

// ---------------------------------------------------------------------------
// K1: projection GEMM  O[m,n] = sum_d X[m,d] * Wt[n,d] + bias[n]
// Block = 256 threads = 8 waves; block tile 32(M) x 256(N); wave tile 16x64.
// Ping-pong double buffering with in-place refills: consume buffer A (k),
// refill A (k+64), consume buffer B (k+32), refill B (k+96) -> no rotating
// register copies, and every WMMA has the partner buffer's loads in flight.
// ---------------------------------------------------------------------------
__global__ void __launch_bounds__(256) proj_gemm_kernel(
    const unsigned short* __restrict__ X,    // [B*T, Dc] bf16
    const unsigned short* __restrict__ Wt,   // [Kc, Dc]  bf16 (W transposed)
    const float* __restrict__ bias,          // [Kc]
    unsigned short* __restrict__ Out,
    int storeTransposed)
{
  const int lane = threadIdx.x & 31;
  const int wave = threadIdx.x >> 5;
  const int m0 = blockIdx.y * 32 + (wave & 1) * 16;
  const int n0 = blockIdx.x * 256 + (wave >> 1) * 64;
  const int row = lane & 15;
  const int mb  = (lane >> 4) * 8;

  const unsigned short* aRow = X + (size_t)(m0 + row) * Dc;
  const unsigned short* bRow[4];
#pragma unroll
  for (int s = 0; s < 4; ++s)
    bRow[s] = Wt + (size_t)(n0 + s * 16 + row) * Dc;

  v8f acc[4] = {};
  v16bf aA = load_frag16(aRow + 0, lane);
  v16bf aB = load_frag16(aRow + 32, lane);
  v16bf bA[4], bB[4];
#pragma unroll
  for (int s = 0; s < 4; ++s) {
    bA[s] = load_frag16(bRow[s] + 0, lane);
    bB[s] = load_frag16(bRow[s] + 32, lane);
  }

  for (int kk = 0; kk < Dc; kk += 64) {
    // consume buffer A (k = kk), refill A for k = kk+64
#pragma unroll
    for (int s = 0; s < 4; ++s) acc[s] = wmma_bf16(aA, bA[s], acc[s]);
    const int k2 = (kk + 64 < Dc) ? kk + 64 : 0;   // dummy wrap on tail
    aA = load_frag16(aRow + k2, lane);
#pragma unroll
    for (int s = 0; s < 4; ++s) bA[s] = load_frag16(bRow[s] + k2, lane);
    // consume buffer B (k = kk+32), refill B for k = kk+96
#pragma unroll
    for (int s = 0; s < 4; ++s) acc[s] = wmma_bf16(aB, bB[s], acc[s]);
    const int k3 = (kk + 96 < Dc) ? kk + 96 : 0;
    aB = load_frag16(aRow + k3, lane);
#pragma unroll
    for (int s = 0; s < 4; ++s) bB[s] = load_frag16(bRow[s] + k3, lane);
  }

#pragma unroll
  for (int s = 0; s < 4; ++s) {
    const int n = n0 + s * 16 + row;
    const float bv = bias[n];
#pragma unroll
    for (int r = 0; r < 8; ++r) {
      float v = acc[s][r] + bv;
      int m = m0 + mb + r;
      if (!storeTransposed) {
        Out[(size_t)m * Kc + n] = f2bf(v);
      } else {
        int b = m / Tc, t = m % Tc;
        Out[((size_t)b * Kc + n) * Tc + t] = f2bf(v);
      }
    }
  }
}

// ---------------------------------------------------------------------------
// K2: column softmax stats.  For every key column j (per batch):
//   m_j = max_{i>=j} q_i.k_j ,  s_j = sum_{i>=j} exp(l - m_j)
// One wave owns 16 columns; paired i-tiles share one K-fragment (2 WMMA per
// 3 fragment loads); ping-pong double-buffered k-loop.
// ---------------------------------------------------------------------------
__global__ void __launch_bounds__(128) col_stats_kernel(
    const unsigned short* __restrict__ Q,   // [B*T, Kc] bf16
    const unsigned short* __restrict__ Kh,  // [B*T, Kc] bf16
    float* __restrict__ colmax, float* __restrict__ colsum)
{
  const float NEG_INF = -__builtin_inff();
  const int b    = blockIdx.y;
  const int lane = threadIdx.x & 31;
  const int wave = threadIdx.x >> 5;
  const int j0   = blockIdx.x * 64 + wave * 16;
  const int row  = lane & 15;
  const int mb   = (lane >> 4) * 8;

  const unsigned short* Qb = Q  + (size_t)b * Tc * Kc;
  const unsigned short* Kb = Kh + (size_t)b * Tc * Kc;
  const int jg = j0 + row;
  const unsigned short* bRow = Kb + (size_t)jg * Kc;

  float m = NEG_INF, s = 0.f;
  const int istart = j0 & ~31;                   // pair-aligned causal start

  for (int i0 = istart; i0 < Tc; i0 += 32) {
    const unsigned short* aRow0 = Qb + (size_t)(i0 + row) * Kc;
    const unsigned short* aRow1 = aRow0 + (size_t)16 * Kc;
    if (i0 + 32 < Tc) {                          // prefetch next i-pair Q rows
      __builtin_prefetch(aRow0 + (size_t)32 * Kc, 0, 3);
      __builtin_prefetch(aRow0 + (size_t)48 * Kc, 0, 3);
    }
    v8f acc0 = {}, acc1 = {};
    v16bf bAb  = load_frag16(bRow, lane);
    v16bf a0Ab = load_frag16(aRow0, lane);
    v16bf a1Ab = load_frag16(aRow1, lane);
    v16bf bBb  = load_frag16(bRow + 32, lane);
    v16bf a0Bb = load_frag16(aRow0 + 32, lane);
    v16bf a1Bb = load_frag16(aRow1 + 32, lane);
    for (int kk = 0; kk < Kc; kk += 64) {
      acc0 = wmma_bf16(a0Ab, bAb, acc0);
      acc1 = wmma_bf16(a1Ab, bAb, acc1);
      const int k2 = (kk + 64 < Kc) ? kk + 64 : 0;
      bAb  = load_frag16(bRow + k2, lane);
      a0Ab = load_frag16(aRow0 + k2, lane);
      a1Ab = load_frag16(aRow1 + k2, lane);
      acc0 = wmma_bf16(a0Bb, bBb, acc0);
      acc1 = wmma_bf16(a1Bb, bBb, acc1);
      const int k3 = (kk + 96 < Kc) ? kk + 96 : 0;
      bBb  = load_frag16(bRow + k3, lane);
      a0Bb = load_frag16(aRow0 + k3, lane);
      a1Bb = load_frag16(aRow1 + k3, lane);
    }

#pragma unroll
    for (int half = 0; half < 2; ++half) {
      const v8f acc = half ? acc1 : acc0;
      const int ib = i0 + half * 16;
      float lmax = NEG_INF;
#pragma unroll
      for (int r = 0; r < 8; ++r)
        if (jg <= ib + mb + r) lmax = fmaxf(lmax, acc[r]);
      float tmax = fmaxf(lmax, __shfl_xor(lmax, 16, 32));
      float lsum = 0.f;
      if (tmax > NEG_INF) {
#pragma unroll
        for (int r = 0; r < 8; ++r)
          if (jg <= ib + mb + r) lsum += __expf(acc[r] - tmax);
      }
      float tsum = lsum + __shfl_xor(lsum, 16, 32);
      if (tmax > NEG_INF) {
        float nm = fmaxf(m, tmax);
        s = s * __expf(m - nm) + tsum * __expf(tmax - nm);
        m = nm;
      }
    }
  }
  if (lane < 16) {
    colmax[(size_t)b * Tc + jg] = m;
    colsum[(size_t)b * Tc + jg] = s;
  }
}

// ---------------------------------------------------------------------------
// K3: out[b,i,v] = x[b,i,v] + sum_{j<=i} exp(l[i,j]-m_j)/(s_j*sqrt(k)) * V[j,v]
// Block = 256 threads = 8 waves; one block covers the FULL 1024 v-columns of a
// 16-row tile, so logits are computed exactly once (no cross-block redundancy).
// The block's 16x1024 Q tile is staged into LDS once via async-to-LDS copies
// (ASYNCcnt).  Per 16x32 P-chunk: 8 waves split the K=1024 logits reduction
// (partials in LDS), column-softmax-scale to bf16 P in LDS, then each wave
// WMMAs P against its 128 columns of V^T.
// ---------------------------------------------------------------------------
__global__ void __launch_bounds__(256) attn_out_kernel(
    const unsigned short* __restrict__ Q,
    const unsigned short* __restrict__ Kh,
    const unsigned short* __restrict__ Vt,       // [B][Dc][Tc] bf16
    const float* __restrict__ colmax,
    const float* __restrict__ colsum,
    const float* __restrict__ Xin,
    float* __restrict__ Out)
{
  __shared__ unsigned short Qs[16][Kc];          // 32 KB staged Q tile
  __shared__ float Lp[8][2][16][16];             // 16 KB partial logits
  __shared__ unsigned short Pb[16][32];          // 1 KB bf16 P tile

  const int i0   = blockIdx.x * 16;
  const int b    = blockIdx.y;
  const int lane = threadIdx.x & 31;
  const int wave = threadIdx.x >> 5;
  const int v0   = wave * 128;
  const int row  = lane & 15;
  const int mb   = (lane >> 4) * 8;

  const unsigned short* Qb = Q  + (size_t)b * Tc * Kc;
  const unsigned short* Kb = Kh + (size_t)b * Tc * Kc;
  const unsigned short* Vb = Vt + (size_t)b * Dc * Tc;
  const float* cm = colmax + (size_t)b * Tc;
  const float* cs = colsum + (size_t)b * Tc;

  // ---- stage the 16x1024 Q tile into LDS with async copies (32 KB) ----
  {
    const char* gsrc = (const char*)(Qb + (size_t)i0 * Kc);
    char* ldst = (char*)&Qs[0][0];
#pragma unroll
    for (int it = 0; it < 8; ++it) {
      const int off = (threadIdx.x + it * 256) * 16;
      async_copy_b128(ldst + off, gsrc + off);
    }
    wait_asynccnt0();
  }
  __syncthreads();

  v8f acc[8] = {};
  const int jmax = i0 + 16;                      // causal: j <= i < i0+16
  const int kbeg = wave * 128;                   // this wave's K-slice

  for (int j0 = 0; j0 < jmax; j0 += 32) {
    // (1) partial logits over this wave's K-range; A (Q) comes from LDS and is
    //     shared by both 16-wide j sub-tiles.
    const unsigned short* bRow0 = Kb + (size_t)(j0 + row) * Kc + kbeg;
    const unsigned short* bRow1 = bRow0 + (size_t)16 * Kc;
    if (j0 + 32 < jmax) {                        // prefetch next chunk's K rows
      __builtin_prefetch(bRow0 + (size_t)32 * Kc, 0, 3);
      __builtin_prefetch(bRow0 + (size_t)48 * Kc, 0, 3);
    }
    v8f p0 = {}, p1 = {};
#pragma unroll
    for (int kk = 0; kk < 128; kk += 32) {
      v16bf a = load_frag16(&Qs[row][kbeg + kk], lane);
      p0 = wmma_bf16(a, load_frag16(bRow0 + kk, lane), p0);
      p1 = wmma_bf16(a, load_frag16(bRow1 + kk, lane), p1);
    }
#pragma unroll
    for (int r = 0; r < 8; ++r) {
      Lp[wave][0][mb + r][row] = p0[r];
      Lp[wave][1][mb + r][row] = p1[r];
    }
    __syncthreads();

    // (2) reduce partials, column-softmax scale (1/sqrt(1024)=1/32 applied
    //     AFTER softmax, per the reference), causal mask, cast to bf16
    for (int e = threadIdx.x; e < 512; e += 256) {
      int jj = e & 31, mm = e >> 5;
      int jc = jj >> 4, nn = jj & 15;
      float l = 0.f;
#pragma unroll
      for (int w = 0; w < 8; ++w) l += Lp[w][jc][mm][nn];
      int gj = j0 + jj, gi = i0 + mm;
      float p = 0.f;
      if (gj <= gi) p = __expf(l - cm[gj]) / (cs[gj] * 32.0f);
      Pb[mm][jj] = f2bf(p);
    }
    __syncthreads();

    // (3) P @ V^T : shared A-fragment from LDS, B-fragments contiguous in Vt
    v16bf a = load_frag16(&Pb[row][0], lane);
#pragma unroll
    for (int s = 0; s < 8; ++s) {
      const unsigned short* bRow = Vb + (size_t)(v0 + s * 16 + row) * Tc + j0;
      __builtin_prefetch(bRow + 32, 0, 3);       // next chunk's segment
      acc[s] = wmma_bf16(a, load_frag16(bRow, lane), acc[s]);
    }
    __syncthreads();
  }

  // epilogue: residual add in fp32
#pragma unroll
  for (int s = 0; s < 8; ++s) {
    const int n = v0 + s * 16 + row;
#pragma unroll
    for (int r = 0; r < 8; ++r) {
      size_t idx = ((size_t)b * Tc + (i0 + mb + r)) * Dc + n;
      Out[idx] = Xin[idx] + acc[s][r];
    }
  }
}

// ---------------------------------------------------------------------------
// Host launcher.  Workspace layout (~141 MB):
//   Xh[B*T*D] bf16, Wqt/Wkt/Wvt[D*K] bf16, Qh/Kh[B*T*K] bf16,
//   Vt[B*D*T] bf16, colmax/colsum[B*T] f32.
// ---------------------------------------------------------------------------
extern "C" void kernel_launch(void* const* d_in, const int* in_sizes, int n_in,
                              void* d_out, int out_size, void* d_ws, size_t ws_size,
                              hipStream_t stream) {
  (void)in_sizes; (void)n_in; (void)out_size; (void)ws_size;
  const float* x  = (const float*)d_in[0];
  const float* Wq = (const float*)d_in[1];
  const float* bq = (const float*)d_in[2];
  const float* Wk = (const float*)d_in[3];
  const float* bk = (const float*)d_in[4];
  const float* Wv = (const float*)d_in[5];
  const float* bv = (const float*)d_in[6];
  float* out = (float*)d_out;

  char* ws = (char*)d_ws;
  size_t off = 0;
  auto take = [&](size_t bytes) -> char* {
    char* p = ws + off;
    off = (off + bytes + 255) & ~(size_t)255;
    return p;
  };
  const size_t MT = (size_t)Bc * Tc;                         // 16384 rows
  unsigned short* Xh   = (unsigned short*)take(MT * Dc * 2);
  unsigned short* Wqt  = (unsigned short*)take((size_t)Dc * Kc * 2);
  unsigned short* Wkt  = (unsigned short*)take((size_t)Dc * Kc * 2);
  unsigned short* Wvt  = (unsigned short*)take((size_t)Dc * Kc * 2);
  unsigned short* Qh   = (unsigned short*)take(MT * Kc * 2);
  unsigned short* Kh2  = (unsigned short*)take(MT * Kc * 2);
  unsigned short* Vt   = (unsigned short*)take(MT * Dc * 2);
  float*          cmax = (float*)take(MT * 4);
  float*          csum = (float*)take(MT * 4);

  const int nX = (int)(MT * Dc);
  cast_bf16_kernel<<<nX / 256, 256, 0, stream>>>(x, Xh, nX);
  transpose_cast_kernel<<<(Dc * Kc) / 256, 256, 0, stream>>>(Wq, Wqt);
  transpose_cast_kernel<<<(Dc * Kc) / 256, 256, 0, stream>>>(Wk, Wkt);
  transpose_cast_kernel<<<(Dc * Kc) / 256, 256, 0, stream>>>(Wv, Wvt);

  dim3 gGemm(Kc / 256, (unsigned)(MT / 32));
  proj_gemm_kernel<<<gGemm, 256, 0, stream>>>(Xh, Wqt, bq, Qh,  0);
  proj_gemm_kernel<<<gGemm, 256, 0, stream>>>(Xh, Wkt, bk, Kh2, 0);
  proj_gemm_kernel<<<gGemm, 256, 0, stream>>>(Xh, Wvt, bv, Vt,  1);

  col_stats_kernel<<<dim3(Tc / 64, Bc), 128, 0, stream>>>(Qh, Kh2, cmax, csum);

  attn_out_kernel<<<dim3(Tc / 16, Bc), 256, 0, stream>>>(
      Qh, Kh2, Vt, cmax, csum, x, out);
}
